// Model_54838142435693
// MI455X (gfx1250) — matrix operand order repaired and checked by reading, not compiled, not run on hardware
//
#include <hip/hip_runtime.h>
#include <hip/hip_bf16.h>

// ---------------------------------------------------------------------------
// BiLSTM-CRF tagger for MI455X (gfx1250, wave32, WMMA).
// S=2048 seq, E=512 embed, H=256 per dir, T=32 tags.
// ---------------------------------------------------------------------------

typedef __attribute__((ext_vector_type(16))) __bf16 v16bf;
typedef __attribute__((ext_vector_type(8)))  __bf16 v8bf;
typedef __attribute__((ext_vector_type(8)))  float  v8f;

static constexpr int S_LEN = 2048;
static constexpr int E_DIM = 512;
static constexpr int H_DIM = 256;
static constexpr int T_TAG = 32;
static constexpr int G4H   = 1024;   // 4*H gates per direction
static constexpr int NCOMB = 2048;   // both directions' gates
static constexpr int TAG_START = 30;
static constexpr int TAG_STOP  = 31;
static constexpr float NEG = -10000.0f;

__device__ __forceinline__ __bf16 f2bf(float f) {
    union { float f; unsigned u; } v; v.f = f;
    unsigned r = v.u + 0x7FFFu + ((v.u >> 16) & 1u);   // round to nearest even
    unsigned short h = (unsigned short)(r >> 16);
    return __builtin_bit_cast(__bf16, h);
}

__device__ __forceinline__ float sigmoidf(float x) {
    return 1.0f / (1.0f + __expf(-x));
}

// Load one A-fragment (16x32 bf16, contiguous k = 16h + s) for this lane.
__device__ __forceinline__ v16bf load_a_frag(const __bf16* __restrict__ p) {
    v8bf alo = *(const v8bf*)(p);
    v8bf ahi = *(const v8bf*)(p + 8);
    return __builtin_shufflevector(alo, ahi,
             0,1,2,3,4,5,6,7,8,9,10,11,12,13,14,15);
}

// ---------------------------------------------------------------------------
// Prep kernels: embedding gather + fp32 -> bf16 weight conversion
// ---------------------------------------------------------------------------
__global__ void gather_embed_kernel(const int* __restrict__ sent,
                                    const float* __restrict__ embed,
                                    __bf16* __restrict__ out) {
    int idx = blockIdx.x * blockDim.x + threadIdx.x;
    if (idx >= S_LEN * E_DIM) return;
    int s = idx >> 9;          // /512
    int e = idx & 511;
    out[idx] = f2bf(embed[(size_t)sent[s] * E_DIM + e]);
}

__global__ void convert_bf16_kernel(const float* __restrict__ src,
                                    __bf16* __restrict__ dst, int n) {
    int idx = blockIdx.x * blockDim.x + threadIdx.x;
    if (idx < n) dst[idx] = f2bf(src[idx]);
}

// ---------------------------------------------------------------------------
// Input-projection GEMM: px[2048,2048] = emb_bf16[2048,512] x Wcomb^T + bias
// Wcomb rows 0..1023 = w_ih_f, 1024..2047 = w_ih_b.
// Each wave computes a 16x64 strip (1 M-tile x 4 N-tiles): the A fragment is
// reused across 4 WMMAs per K-chunk.  8 waves/block, 4096 wave strips.
// ---------------------------------------------------------------------------
__global__ void gemm_px_kernel(const __bf16* __restrict__ A,
                               const __bf16* __restrict__ W,
                               const float* __restrict__ bf,
                               const float* __restrict__ bb,
                               float* __restrict__ C) {
    const int lane = threadIdx.x & 31;
    const int id   = blockIdx.x * 8 + (threadIdx.x >> 5);
    const int tm   = id >> 5;            // 128 M tiles
    const int tq   = id & 31;            // 32 N quads (x4 tiles = 128 N tiles)
    const int mr   = lane & 15;
    const int hbA  = (lane < 16) ? 0 : 16;
    const int hbB  = (lane < 16) ? 0 : 16;

    const __bf16* arow = A + (size_t)(tm * 16 + mr) * E_DIM + hbA;
    const __bf16* wrow = W + (size_t)(tq * 64 + mr) * E_DIM + hbB;

    v8f acc[4] = {};
    for (int kb = 0; kb < E_DIM; kb += 32) {
        v16bf av = load_a_frag(arow + kb);
#pragma unroll
        for (int j = 0; j < 4; ++j) {
            v16bf bv = *(const v16bf*)(wrow + (size_t)(j * 16) * E_DIM + kb);
            acc[j] = __builtin_amdgcn_wmma_f32_16x16x32_bf16(
                         false, av, false, bv, (short)0, acc[j], false, false);
        }
    }
    const int mbase = tm * 16 + ((lane < 16) ? 0 : 8);
#pragma unroll
    for (int j = 0; j < 4; ++j) {
        const int n = tq * 64 + j * 16 + mr;
        const float bias = (n < G4H) ? bf[n] : bb[n - G4H];
#pragma unroll
        for (int r = 0; r < 8; ++r)
            C[(size_t)(mbase + r) * NCOMB + n] = acc[j][r] + bias;
    }
}

// ---------------------------------------------------------------------------
// Recurrent LSTM.  grid.x = 2 (dir), block = 1024 threads = 32 waves.
// w_hh (bf16, [1024,256]) lives in VGPRs as persistent B fragments:
//   wave w owns gate tiles n0=2w, n1=2w+1; 8 K-chunks each -> 128 VGPRs/lane.
// Per step: A = h broadcast to all 16 rows (from LDS), 16 chained WMMAs,
// lanes 0-15 emit tile n0, lanes 16-31 emit tile n1 -> g[1024] in LDS,
// then threads 0..255 run the gate nonlinearity and update c/h, prefetching
// the next step's px row (global_prefetch_b8) to hide L2/HBM latency.
// ---------------------------------------------------------------------------
__global__ void __launch_bounds__(1024, 1)
lstm_kernel(const float* __restrict__ px,        // [S, 2048]
            const __bf16* __restrict__ whh_all,  // [2][1024,256] bf16
            const float* __restrict__ h0,        // [2, H]
            const float* __restrict__ c0,        // [2, H]
            __bf16* __restrict__ hs) {           // [S, 2H] bf16 (fwd|bwd)
    const int dir  = blockIdx.x;
    const int tid  = threadIdx.x;
    const int wave = tid >> 5;
    const int lane = tid & 31;

    __shared__ __align__(32) __bf16 h_sh[H_DIM];
    __shared__ float g_sh[G4H];

    const __bf16* whh = whh_all + (size_t)dir * G4H * H_DIM;

    // Persistent B fragments for gate tiles n0 = 2*wave, n1 = 2*wave+1.
    const int nr  = (lane & 15);
    const int hbB = (lane < 16) ? 0 : 16;
    v16bf B0[8], B1[8];
#pragma unroll
    for (int kt = 0; kt < 8; ++kt) {
        B0[kt] = *(const v16bf*)(whh + (size_t)((2 * wave    ) * 16 + nr) * H_DIM
                                 + kt * 32 + hbB);
        B1[kt] = *(const v16bf*)(whh + (size_t)((2 * wave + 1) * 16 + nr) * H_DIM
                                 + kt * 32 + hbB);
    }

    // State init
    float c_reg = 0.0f;
    if (tid < H_DIM) {
        c_reg = c0[dir * H_DIM + tid];
        h_sh[tid] = f2bf(h0[dir * H_DIM + tid]);
    }
    __syncthreads();

    const int hbA = (lane < 16) ? 0 : 16;

    for (int t = 0; t < S_LEN; ++t) {
        const int row = dir ? (S_LEN - 1 - t) : t;

        // g = h @ w_hh^T  (all 16 A-rows replicated to h -> every acc row valid)
        v8f acc0 = {}, acc1 = {};
#pragma unroll
        for (int kt = 0; kt < 8; ++kt) {
            v16bf av = load_a_frag(h_sh + kt * 32 + hbA);
            acc0 = __builtin_amdgcn_wmma_f32_16x16x32_bf16(
                       false, av, false, B0[kt], (short)0, acc0, false, false);
            acc1 = __builtin_amdgcn_wmma_f32_16x16x32_bf16(
                       false, av, false, B1[kt], (short)0, acc1, false, false);
        }
        if (lane < 16) g_sh[(2 * wave)     * 16 + lane]        = acc0[0];
        else           g_sh[(2 * wave + 1) * 16 + (lane - 16)] = acc1[0];
        __syncthreads();

        // Gate nonlinearity + state update on threads 0..255
        if (tid < H_DIM) {
            const float* p = px + (size_t)row * NCOMB + dir * G4H;
            float ig = g_sh[tid]             + p[tid];
            float fg = g_sh[H_DIM + tid]     + p[H_DIM + tid];
            float gg = g_sh[2 * H_DIM + tid] + p[2 * H_DIM + tid];
            float og = g_sh[3 * H_DIM + tid] + p[3 * H_DIM + tid];
            c_reg = sigmoidf(fg) * c_reg + sigmoidf(ig) * tanhf(gg);
            float hval = sigmoidf(og) * tanhf(c_reg);
            __bf16 hb = f2bf(hval);
            h_sh[tid] = hb;
            hs[(size_t)row * (2 * H_DIM) + dir * H_DIM + tid] = hb;

            // Prefetch next step's px row into cache (global_prefetch_b8)
            if (t + 1 < S_LEN) {
                const int nrow = dir ? (S_LEN - 2 - t) : (t + 1);
                const float* pn = px + (size_t)nrow * NCOMB + dir * G4H;
                __builtin_prefetch(pn + tid, 0, 3);
                __builtin_prefetch(pn + H_DIM + tid, 0, 3);
                __builtin_prefetch(pn + 2 * H_DIM + tid, 0, 3);
                __builtin_prefetch(pn + 3 * H_DIM + tid, 0, 3);
            }
        }
        __syncthreads();
    }
}

// ---------------------------------------------------------------------------
// Tag projection: feats[2048,32] = hs[2048,512] x w_tag^T + b_tag
// Each wave computes one 16x32 strip (both N tiles), reusing the A fragment.
// 128 wave strips, 8 waves/block -> 16 blocks.
// ---------------------------------------------------------------------------
__global__ void gemm_tag_kernel(const __bf16* __restrict__ A,
                                const __bf16* __restrict__ W,
                                const float* __restrict__ bt,
                                float* __restrict__ C) {
    const int lane = threadIdx.x & 31;
    const int tm   = blockIdx.x * 8 + (threadIdx.x >> 5);   // 128 M tiles
    const int mr   = lane & 15;
    const int hbA  = (lane < 16) ? 0 : 16;
    const int hbB  = (lane < 16) ? 0 : 16;
    const int K    = 2 * H_DIM;

    const __bf16* arow = A + (size_t)(tm * 16 + mr) * K + hbA;
    const __bf16* wrow = W + (size_t)mr * K + hbB;

    v8f acc[2] = {};
    for (int kb = 0; kb < K; kb += 32) {
        v16bf av = load_a_frag(arow + kb);
#pragma unroll
        for (int j = 0; j < 2; ++j) {
            v16bf bv = *(const v16bf*)(wrow + (size_t)(j * 16) * K + kb);
            acc[j] = __builtin_amdgcn_wmma_f32_16x16x32_bf16(
                         false, av, false, bv, (short)0, acc[j], false, false);
        }
    }
    const int mbase = tm * 16 + ((lane < 16) ? 0 : 8);
#pragma unroll
    for (int j = 0; j < 2; ++j) {
        const int n = j * 16 + mr;
        const float bias = bt[n];
#pragma unroll
        for (int r = 0; r < 8; ++r)
            C[(size_t)(mbase + r) * T_TAG + n] = acc[j][r] + bias;
    }
}

// ---------------------------------------------------------------------------
// Viterbi: single wave32, lane = next-tag.  Transition row in 32 VGPRs.
// ---------------------------------------------------------------------------
__global__ void viterbi_kernel(const float* __restrict__ feats,  // [S, 32]
                               const float* __restrict__ trans,  // [32, 32]
                               int* __restrict__ bptrs,          // [S, 32]
                               float* __restrict__ out) {        // [1 + S]
    const int lane = threadIdx.x;
    __shared__ float fv[T_TAG];
    __shared__ float term_sh[T_TAG];

    float tr[T_TAG];
#pragma unroll
    for (int p = 0; p < T_TAG; ++p) tr[p] = trans[lane * T_TAG + p];

    fv[lane] = (lane == TAG_START) ? 0.0f : NEG;
    __syncthreads();

    for (int t = 0; t < S_LEN; ++t) {
        float best = -3.4e38f; int bi = 0;
#pragma unroll
        for (int p = 0; p < T_TAG; ++p) {
            float s = fv[p] + tr[p];
            if (s > best) { best = s; bi = p; }
        }
        float nf = best + feats[t * T_TAG + lane];
        bptrs[t * T_TAG + lane] = bi;
        __syncthreads();
        fv[lane] = nf;
        __syncthreads();
    }

    term_sh[lane] = fv[lane] + trans[TAG_STOP * T_TAG + lane];
    __syncthreads();
    if (lane == 0) {
        float best = term_sh[0]; int bi = 0;
        for (int p = 1; p < T_TAG; ++p)
            if (term_sh[p] > best) { best = term_sh[p]; bi = p; }
        out[0] = best;
        int tag = bi;
        for (int t = S_LEN - 1; t >= 0; --t) {
            out[1 + t] = (float)tag;
            tag = bptrs[t * T_TAG + tag];
        }
    }
}

// ---------------------------------------------------------------------------
// Host-side launch
// ---------------------------------------------------------------------------
extern "C" void kernel_launch(void* const* d_in, const int* in_sizes, int n_in,
                              void* d_out, int out_size, void* d_ws, size_t ws_size,
                              hipStream_t stream) {
    (void)in_sizes; (void)n_in; (void)out_size; (void)ws_size;

    const int*   sentence = (const int*)  d_in[0];
    const float* embed    = (const float*)d_in[1];
    const float* w_ih_f   = (const float*)d_in[2];
    const float* w_hh_f   = (const float*)d_in[3];
    const float* b_f      = (const float*)d_in[4];
    const float* w_ih_b   = (const float*)d_in[5];
    const float* w_hh_b   = (const float*)d_in[6];
    const float* b_b      = (const float*)d_in[7];
    const float* w_tag    = (const float*)d_in[8];
    const float* b_tag    = (const float*)d_in[9];
    const float* trans    = (const float*)d_in[10];
    const float* h0       = (const float*)d_in[11];
    const float* c0       = (const float*)d_in[12];
    float* out = (float*)d_out;

    // Workspace layout (all offsets 256B-aligned)
    char* ws = (char*)d_ws;
    size_t off = 0;
    auto alloc = [&](size_t bytes) { char* p = ws + off; off += (bytes + 255) & ~size_t(255); return p; };
    __bf16* emb_bf  = (__bf16*)alloc((size_t)S_LEN * E_DIM * 2);      // [2048,512]
    __bf16* wih_bf  = (__bf16*)alloc((size_t)NCOMB * E_DIM * 2);      // [2048,512]
    __bf16* whh_bf  = (__bf16*)alloc((size_t)2 * G4H * H_DIM * 2);    // [2][1024,256]
    __bf16* wtag_bf = (__bf16*)alloc((size_t)T_TAG * 2 * H_DIM * 2);  // [32,512]
    float*  px      = (float*) alloc((size_t)S_LEN * NCOMB * 4);      // [2048,2048]
    __bf16* hs_bf   = (__bf16*)alloc((size_t)S_LEN * 2 * H_DIM * 2);  // [2048,512]
    float*  feats   = (float*) alloc((size_t)S_LEN * T_TAG * 4);      // [2048,32]
    int*    bptrs   = (int*)   alloc((size_t)S_LEN * T_TAG * 4);      // [2048,32]

    // 1) Embedding gather + bf16 conversions
    gather_embed_kernel<<<(S_LEN * E_DIM + 255) / 256, 256, 0, stream>>>(
        sentence, embed, emb_bf);
    convert_bf16_kernel<<<(G4H * E_DIM + 255) / 256, 256, 0, stream>>>(
        w_ih_f, wih_bf, G4H * E_DIM);
    convert_bf16_kernel<<<(G4H * E_DIM + 255) / 256, 256, 0, stream>>>(
        w_ih_b, wih_bf + (size_t)G4H * E_DIM, G4H * E_DIM);
    convert_bf16_kernel<<<(G4H * H_DIM + 255) / 256, 256, 0, stream>>>(
        w_hh_f, whh_bf, G4H * H_DIM);
    convert_bf16_kernel<<<(G4H * H_DIM + 255) / 256, 256, 0, stream>>>(
        w_hh_b, whh_bf + (size_t)G4H * H_DIM, G4H * H_DIM);
    convert_bf16_kernel<<<(T_TAG * 2 * H_DIM + 255) / 256, 256, 0, stream>>>(
        w_tag, wtag_bf, T_TAG * 2 * H_DIM);

    // 2) Input-projection GEMM: px = emb x [w_ih_f;w_ih_b]^T + bias
    //    4096 wave strips (16x64 each), 8 waves/block
    gemm_px_kernel<<<512, 256, 0, stream>>>(emb_bf, wih_bf, b_f, b_b, px);

    // 3) Recurrent BiLSTM (2 workgroups, one per direction, w_hh in VGPRs)
    lstm_kernel<<<2, 1024, 0, stream>>>(px, whh_bf, h0, c0, hs_bf);

    // 4) Tag projection GEMM
    gemm_tag_kernel<<<16, 256, 0, stream>>>(hs_bf, wtag_bf, b_tag, feats);

    // 5) Viterbi decode (single wave)
    viterbi_kernel<<<1, 32, 0, stream>>>(feats, trans, bptrs, out);
}
